// GraphSAGEClassifier_41841571397708
// MI455X (gfx1250) — matrix operand order repaired and checked by reading, not compiled, not run on hardware
//
#include <hip/hip_runtime.h>

typedef _Float16 f16;
typedef __attribute__((ext_vector_type(16))) _Float16 v16h;
typedef __attribute__((ext_vector_type(8)))  _Float16 v8h;
typedef __attribute__((ext_vector_type(8)))  float    v8f;

#define CH 128  // IN_CH == HID == 128

// ---------------------------------------------------------------- zero fill
__global__ void zero_f32(float* __restrict__ p, long long n) {
    long long i = (long long)blockIdx.x * blockDim.x + threadIdx.x;
    if (i < n) p[i] = 0.0f;
}

// ---------------------------------------------------------------- degree
__global__ void deg_kernel(const int* __restrict__ ei, float* __restrict__ deg, int E) {
    int i = blockIdx.x * blockDim.x + threadIdx.x;
    if (i < E) atomicAdd(deg + ei[E + i], 1.0f);   // ei[1][e] = dst
}

// ---------------------------------------------------------------- scatter-add of source rows
// one wave (32 lanes) per edge, 4 channels per lane (float4 row read, L2-resident atomics)
__global__ __launch_bounds__(256) void scatter_kernel(const float* __restrict__ H,
                                                      const int* __restrict__ ei,
                                                      float* __restrict__ agg, int E) {
    int wid  = threadIdx.x >> 5;
    int lane = threadIdx.x & 31;
    int e = blockIdx.x * 8 + wid;
    if (e >= E) return;
    int src = ei[e];
    int dst = ei[E + e];
    const float4 v = *(const float4*)(H + (size_t)src * CH + lane * 4);
    float* a = agg + (size_t)dst * CH + lane * 4;
    atomicAdd(a + 0, v.x);
    atomicAdd(a + 1, v.y);
    atomicAdd(a + 2, v.z);
    atomicAdd(a + 3, v.w);
}

// ---------------------------------------------------------------- weight -> WMMA B-fragment order
// B (32x16 f16) layout per ISA 7.12.2: lane L, elem i -> K = kb + 8*(L/16) + i + (i>=8 ? 8 : 0), N = ntile*16 + L%16
// Flat: frag[((ntile*4 + kstep)*32 + lane)*16 + i]  => one coalesced v16h load per wave/kstep.
__global__ void prep_weights(const float* __restrict__ W, f16* __restrict__ out) {
    int t = blockIdx.x * blockDim.x + threadIdx.x;   // 0..1023
    int nt   = t >> 7;          // 0..7  (N tile)
    int ks   = (t >> 5) & 3;    // 0..3  (K step of 32)
    int lane = t & 31;
    int g = lane >> 4;
    int N = nt * 16 + (lane & 15);
    f16* o = out + (((size_t)(nt * 4 + ks) * 32 + lane) * 16);
#pragma unroll
    for (int i = 0; i < 16; ++i) {
        int K = ks * 32 + 8 * g + i + (i >= 8 ? 8 : 0);
        o[i] = (f16)W[K * CH + N];
    }
}

// ---------------------------------------------------------------- SAGE layer GEMM via WMMA
// out = relu( (agg/deg) @ Wl + H @ Wr + b ),  block = 16-row M tile, 8 waves = 8 N tiles
__global__ __launch_bounds__(256) void sage_gemm(const float* __restrict__ H,
                                                 const float* __restrict__ agg,
                                                 const float* __restrict__ deg,
                                                 const f16* __restrict__ WlF,
                                                 const f16* __restrict__ WrF,
                                                 const float* __restrict__ bias,
                                                 float* __restrict__ out) {
    __shared__ f16 sAgg[16 * CH];
    __shared__ f16 sH[16 * CH];

    const int t  = threadIdx.x;
    const int m0 = blockIdx.x * 16;

    // stage 16x128 tiles into LDS as f16; agg tile is degree-normalized here
    {
        int row = t >> 4;
        int cb  = (t & 15) * 8;
        int gr  = m0 + row;
        float invd = 1.0f / fmaxf(deg[gr], 1.0f);
        const float4* ap = (const float4*)(agg + (size_t)gr * CH + cb);
        const float4* hp = (const float4*)(H   + (size_t)gr * CH + cb);
        float4 a0 = ap[0], a1 = ap[1];
        float4 h0 = hp[0], h1 = hp[1];
        f16* da = sAgg + row * CH + cb;
        f16* dh = sH   + row * CH + cb;
        da[0] = (f16)(a0.x * invd); da[1] = (f16)(a0.y * invd);
        da[2] = (f16)(a0.z * invd); da[3] = (f16)(a0.w * invd);
        da[4] = (f16)(a1.x * invd); da[5] = (f16)(a1.y * invd);
        da[6] = (f16)(a1.z * invd); da[7] = (f16)(a1.w * invd);
        dh[0] = (f16)h0.x; dh[1] = (f16)h0.y; dh[2] = (f16)h0.z; dh[3] = (f16)h0.w;
        dh[4] = (f16)h1.x; dh[5] = (f16)h1.y; dh[6] = (f16)h1.z; dh[7] = (f16)h1.w;
    }
    __syncthreads();

    const int lane = t & 31;
    const int wave = t >> 5;          // N tile index
    const int g    = lane >> 4;       // lane group
    const int M    = lane & 15;       // A row / D column index

    v8f acc;
    float bv = bias[wave * 16 + M];   // D: N = lane%16
#pragma unroll
    for (int r = 0; r < 8; ++r) acc[r] = bv;

    const f16* ra = sAgg + M * CH;
    const f16* rh = sH   + M * CH;

#pragma unroll
    for (int ks = 0; ks < 4; ++ks) {
        const int kb = ks * 32;
        // A fragment (16x32 f16): chunks [kb+8g, +8) and [kb+16+8g, +8)
        v8h loA = *(const v8h*)(ra + kb + 8 * g);
        v8h hiA = *(const v8h*)(ra + kb + 16 + 8 * g);
        v8h loH = *(const v8h*)(rh + kb + 8 * g);
        v8h hiH = *(const v8h*)(rh + kb + 16 + 8 * g);
        v16h aA = __builtin_shufflevector(loA, hiA, 0,1,2,3,4,5,6,7,8,9,10,11,12,13,14,15);
        v16h aH = __builtin_shufflevector(loH, hiH, 0,1,2,3,4,5,6,7,8,9,10,11,12,13,14,15);
        // pre-shuffled B fragments: one coalesced 32B load per matrix
        const v16h bL = *(const v16h*)(WlF + (((size_t)(wave * 4 + ks) * 32 + lane) * 16));
        const v16h bR = *(const v16h*)(WrF + (((size_t)(wave * 4 + ks) * 32 + lane) * 16));
        acc = __builtin_amdgcn_wmma_f32_16x16x32_f16(false, aA, false, bL, (short)0, acc, false, false);
        acc = __builtin_amdgcn_wmma_f32_16x16x32_f16(false, aH, false, bR, (short)0, acc, false, false);
    }

    // ReLU + store: D element r -> row m0 + r + 8g, col wave*16 + lane%16
    float* o = out + (size_t)(m0 + 8 * g) * CH + wave * 16 + M;
#pragma unroll
    for (int r = 0; r < 8; ++r) {
        float v = acc[r];
        o[(size_t)r * CH] = v > 0.0f ? v : 0.0f;
    }
}

// ---------------------------------------------------------------- logits = h2 @ Wout + bout (Wout: 128x2)
__global__ void out_kernel(const float* __restrict__ h2, const float* __restrict__ Wout,
                           const float* __restrict__ bout, float* __restrict__ logits, int n) {
    int i = blockIdx.x * blockDim.x + threadIdx.x;
    if (i >= n) return;
    float a0 = bout[0], a1 = bout[1];
    const float* r = h2 + (size_t)i * CH;
#pragma unroll 8
    for (int c = 0; c < CH; ++c) {
        float v = r[c];
        a0 += v * Wout[2 * c];
        a1 += v * Wout[2 * c + 1];
    }
    logits[2 * i]     = a0;
    logits[2 * i + 1] = a1;
}

// ---------------------------------------------------------------- launcher
extern "C" void kernel_launch(void* const* d_in, const int* in_sizes, int n_in,
                              void* d_out, int out_size, void* d_ws, size_t ws_size,
                              hipStream_t stream) {
    const float* x    = (const float*)d_in[0];
    const int*   ei   = (const int*)d_in[1];   // [2][E] int32
    const float* W1l  = (const float*)d_in[2];
    const float* W1r  = (const float*)d_in[3];
    const float* b1   = (const float*)d_in[4];
    const float* W2l  = (const float*)d_in[5];
    const float* W2r  = (const float*)d_in[6];
    const float* b2   = (const float*)d_in[7];
    const float* Wout = (const float*)d_in[8];
    const float* bout = (const float*)d_in[9];

    const int n = in_sizes[0] / CH;       // 50000 (multiple of 16)
    const int E = in_sizes[1] / 2;        // 800000

    // workspace carve-up
    float* agg = (float*)d_ws;                       // n*128 f32
    float* deg = agg + (size_t)n * CH;               // n f32 (contiguous after agg)
    float* h1  = deg + n;                            // n*128 f32
    f16* w1lf = (f16*)(h1 + (size_t)n * CH);         // 4x 16384 f16 fragment buffers
    f16* w1rf = w1lf + 8 * 4 * 32 * 16;
    f16* w2lf = w1rf + 8 * 4 * 32 * 16;
    f16* w2rf = w2lf + 8 * 4 * 32 * 16;

    float* logits = (float*)d_out;                   // [n,2]
    float* h2     = logits + (size_t)2 * n;          // [n,128] (tuple output #2)

    // pre-shuffle all weights into WMMA B-fragment order (f16)
    prep_weights<<<8, 128, 0, stream>>>(W1l, w1lf);
    prep_weights<<<8, 128, 0, stream>>>(W1r, w1rf);
    prep_weights<<<8, 128, 0, stream>>>(W2l, w2lf);
    prep_weights<<<8, 128, 0, stream>>>(W2r, w2rf);

    // ---- layer 1 ----
    long long zn = (long long)n * CH + n;            // agg + deg together
    zero_f32<<<(int)((zn + 255) / 256), 256, 0, stream>>>(agg, zn);
    deg_kernel<<<(E + 255) / 256, 256, 0, stream>>>(ei, deg, E);
    scatter_kernel<<<(E + 7) / 8, 256, 0, stream>>>(x, ei, agg, E);
    sage_gemm<<<n / 16, 256, 0, stream>>>(x, agg, deg, w1lf, w1rf, b1, h1);

    // ---- layer 2 (same graph -> reuse deg) ----
    long long zn2 = (long long)n * CH;
    zero_f32<<<(int)((zn2 + 255) / 256), 256, 0, stream>>>(agg, zn2);
    scatter_kernel<<<(E + 7) / 8, 256, 0, stream>>>(h1, ei, agg, E);
    sage_gemm<<<n / 16, 256, 0, stream>>>(h1, agg, deg, w2lf, w2rf, b2, h2);

    // ---- output head ----
    out_kernel<<<(n + 255) / 256, 256, 0, stream>>>(h2, Wout, bout, logits, n);
}
